// GCN_QN_1_61546881352330
// MI455X (gfx1250) — matrix operand order repaired and checked by reading, not compiled, not run on hardware
//
#include <hip/hip_runtime.h>
#include <hip/hip_bf16.h>
#include <cstdint>
#include <cstddef>

// ---------------------------------------------------------------------------
// GCN Q-network forward for MI455X (gfx1250, wave32, WMMA).
// Dominant op: per-batch [2048x2048] x [2048x64] pooling GEMM, done with
// v_wmma_f32_16x16x32_bf16; adj streamed f32->bf16 in-register, gv never
// materialized (row scale + identity folded into the epilogue).
// ---------------------------------------------------------------------------

typedef __attribute__((ext_vector_type(16))) __bf16 v16bf;
typedef __attribute__((ext_vector_type(8)))  float  v8f;

constexpr int NB = 16;    // batch
constexpr int NN = 2048;  // nodes
constexpr int EE = 64;    // embedding
constexpr int RG = 32;    // reg width

__device__ __forceinline__ float relu_(float x) { return fmaxf(x, 0.0f); }

// ---- D = column sums of adj; invD = 1/(D==0 ? 0.01 : D) --------------------
__global__ void __launch_bounds__(256)
k_colsum_inv(const float* __restrict__ adj, float* __restrict__ invD) {
  const int b = blockIdx.y;
  const int j = blockIdx.x * 256 + threadIdx.x;
  const float* a = adj + (size_t)b * NN * NN + j;
  float s = 0.0f;
  for (int i = 0; i < NN; ++i) s += a[(size_t)i * NN];
  if (s == 0.0f) s = 0.01f;
  invD[b * NN + j] = 1.0f / s;
}

// ---- mu = relu(xv * mu1) ---------------------------------------------------
__global__ void __launch_bounds__(256)
k_mu_init(const float* __restrict__ xv, const float* __restrict__ mu1,
          float* __restrict__ mu) {
  const int idx = blockIdx.x * 256 + threadIdx.x;   // < NB*NN*EE
  const int e  = idx & (EE - 1);
  const int bn = idx >> 6;
  mu[idx] = relu_(xv[bn] * mu1[e]);
}

// ---- mu_p = relu(mu @ W_pre^T + b); emit f32 [N,E] and bf16 [E,N]^T --------
__global__ void __launch_bounds__(256)
k_pre(const float* __restrict__ mu, const float* __restrict__ W,
      const float* __restrict__ bias, float* __restrict__ mup,
      __bf16* __restrict__ mupT) {
  __shared__ float  WT[EE * EE];     // W transposed: WT[k*EE+e]
  __shared__ float  muS[64 * EE];
  __shared__ __bf16 sT[EE * 64];
  const int tid   = threadIdx.x;
  const int rbase = blockIdx.x * 64;         // global row (b*NN + n)
  const int b     = rbase >> 11;
  const int nloc  = rbase & (NN - 1);
  for (int idx = tid; idx < EE * EE; idx += 256) {
    const int e = idx >> 6, k = idx & 63;
    WT[k * EE + e] = W[idx];
  }
  for (int idx = tid; idx < 64 * EE; idx += 256)
    muS[idx] = mu[(size_t)rbase * EE + idx];
  __syncthreads();
  const int e  = tid & 63;
  const int r0 = tid >> 6;
  const float bb = bias[e];
  for (int p = 0; p < 16; ++p) {
    const int rl = r0 + 4 * p;
    float acc = bb;
#pragma unroll
    for (int k = 0; k < EE; ++k) acc += muS[rl * EE + k] * WT[k * EE + e];
    const float v = relu_(acc);
    mup[(size_t)(rbase + rl) * EE + e] = v;
    sT[e * 64 + rl] = (__bf16)v;
  }
  __syncthreads();
  for (int idx = tid; idx < EE * 64; idx += 256) {
    const int ee = idx >> 6, rl = idx & 63;
    mupT[(size_t)b * EE * NN + (size_t)ee * NN + nloc + rl] = sT[idx];
  }
}

// ---- pool = rowscale(invD) * (adj @ mu_p + mu_p)  via bf16 WMMA ------------
// Block: 256 thr = 8 waves; wave owns 16 rows x 64 cols (4 v8f accumulators).
__global__ void __launch_bounds__(256)
k_pool_wmma(const float* __restrict__ adj, const float* __restrict__ invD,
            const __bf16* __restrict__ mupT, const float* __restrict__ mup,
            float* __restrict__ pool) {
  const int b     = blockIdx.y;
  const int tid   = threadIdx.x;
  const int wv    = tid >> 5;
  const int lane  = tid & 31;
  const int lanem = lane & 15;
  const int hsel  = lane >> 4;
  const int row0  = blockIdx.x * 128 + wv * 16;

  const float*  adjrow = adj  + (size_t)b * NN * NN + (size_t)(row0 + lanem) * NN;
  const __bf16* bT     = mupT + (size_t)b * EE * NN + (size_t)lanem * NN + (hsel << 4);

  v8f acc[4] = {};

#pragma unroll 2
  for (int k0 = 0; k0 < NN; k0 += 32) {
    // A fragment source: 16 f32 from this lane's adj row (contiguous k = 16*hsel + s, same order as B).
    const float* ap = adjrow + k0 + (hsel << 4);
    const float4 f0 = *(const float4*)(ap);
    const float4 f1 = *(const float4*)(ap + 4);
    const float4 f2 = *(const float4*)(ap + 8);
    const float4 f3 = *(const float4*)(ap + 12);
    // B fragments: issue all 8 b128 loads up-front so one wait covers them.
    const __bf16* bp = bT + k0;
    v16bf bf[4];
#pragma unroll
    for (int f = 0; f < 4; ++f)
      bf[f] = *(const v16bf*)(bp + (size_t)f * 16 * NN);
    // Convert A to bf16 while B loads are in flight.
    v16bf a;
    a[0]  = (__bf16)f0.x; a[1]  = (__bf16)f0.y; a[2]  = (__bf16)f0.z; a[3]  = (__bf16)f0.w;
    a[4]  = (__bf16)f1.x; a[5]  = (__bf16)f1.y; a[6]  = (__bf16)f1.z; a[7]  = (__bf16)f1.w;
    a[8]  = (__bf16)f2.x; a[9]  = (__bf16)f2.y; a[10] = (__bf16)f2.z; a[11] = (__bf16)f2.w;
    a[12] = (__bf16)f3.x; a[13] = (__bf16)f3.y; a[14] = (__bf16)f3.z; a[15] = (__bf16)f3.w;
#pragma unroll
    for (int f = 0; f < 4; ++f)
      acc[f] = __builtin_amdgcn_wmma_f32_16x16x32_bf16(
          /*neg_a=*/false, a, /*neg_b=*/false, bf[f],
          /*c_mod=*/(short)0, acc[f], /*reuse_a=*/false, /*reuse_b=*/false);
  }

  // Epilogue: out[i,:] = invD[i] * (acc[i,:] + mu_p[i,:])
  const float* invb  = invD + b * NN;
  const float* mupb  = mup  + (size_t)b * NN * EE;
  float*       poolb = pool + (size_t)b * NN * EE;
  float iv[8];
#pragma unroll
  for (int v = 0; v < 8; ++v) iv[v] = invb[row0 + v + (hsel << 3)];
#pragma unroll
  for (int f = 0; f < 4; ++f) {
    const int col = f * 16 + lanem;
#pragma unroll
    for (int v = 0; v < 8; ++v) {
      const int row = row0 + v + (hsel << 3);
      poolb[(size_t)row * EE + col] =
          iv[v] * (acc[f][v] + mupb[(size_t)row * EE + col]);
    }
  }
}

// ---- mu = relu(relu(xv*mu1) + relu(pool@Wpost^T+b)@Wmu2^T + b2) ------------
__global__ void __launch_bounds__(256)
k_post(const float* __restrict__ pool, const float* __restrict__ xv,
       const float* __restrict__ mu1, const float* __restrict__ Wpost,
       const float* __restrict__ bpost, const float* __restrict__ Wmu2,
       const float* __restrict__ bmu2, float* __restrict__ mu) {
  __shared__ float WT1[EE * EE];
  __shared__ float WT2[EE * EE];
  __shared__ float pS[64 * EE];
  __shared__ float hS[64 * EE];
  const int tid   = threadIdx.x;
  const int rbase = blockIdx.x * 64;
  for (int idx = tid; idx < EE * EE; idx += 256) {
    const int e = idx >> 6, k = idx & 63;
    WT1[k * EE + e] = Wpost[idx];
    WT2[k * EE + e] = Wmu2[idx];
  }
  for (int idx = tid; idx < 64 * EE; idx += 256)
    pS[idx] = pool[(size_t)rbase * EE + idx];
  __syncthreads();
  const int e  = tid & 63;
  const int r0 = tid >> 6;
  const float b1 = bpost[e], b2 = bmu2[e], m1 = mu1[e];
  for (int p = 0; p < 16; ++p) {
    const int rl = r0 + 4 * p;
    float acc = b1;
#pragma unroll
    for (int k = 0; k < EE; ++k) acc += pS[rl * EE + k] * WT1[k * EE + e];
    hS[rl * EE + e] = relu_(acc);
  }
  __syncthreads();
  for (int p = 0; p < 16; ++p) {
    const int rl = r0 + 4 * p;
    float acc = b2;
#pragma unroll
    for (int k = 0; k < EE; ++k) acc += hS[rl * EE + k] * WT2[k * EE + e];
    const float mua = relu_(xv[rbase + rl] * m1);
    mu[(size_t)(rbase + rl) * EE + e] = relu_(mua + acc);
  }
}

// ---- pooled[b,e] = sum_n xv[b,n]*mu[b,n,e] ---------------------------------
__global__ void __launch_bounds__(256)
k_pooled(const float* __restrict__ xv, const float* __restrict__ mu,
         float* __restrict__ pooled) {
  __shared__ float red[256];
  const int b = blockIdx.x, tid = threadIdx.x;
  const int e = tid & 63, c = tid >> 6;
  const float* xb = xv + b * NN;
  const float* mb = mu + (size_t)b * NN * EE;
  float acc = 0.0f;
  for (int n = c; n < NN; n += 4) acc += xb[n] * mb[(size_t)n * EE + e];
  red[tid] = acc;
  __syncthreads();
  if (c == 0)
    pooled[b * EE + e] = red[e] + red[64 + e] + red[128 + e] + red[192 + e];
}

// ---- q1reg[b,r] = breg[r] + sum_j (pooled@Wq1^T + bq1)[j] * Wreg[r,j] ------
__global__ void __launch_bounds__(64)
k_q1reg(const float* __restrict__ pooled, const float* __restrict__ Wq1,
        const float* __restrict__ bq1, const float* __restrict__ Wreg,
        const float* __restrict__ breg, float* __restrict__ q1reg) {
  __shared__ float pv[EE], q1v[EE];
  const int b = blockIdx.x, t = threadIdx.x;
  pv[t] = pooled[b * EE + t];
  __syncthreads();
  float acc = bq1[t];
  for (int k = 0; k < EE; ++k) acc += pv[k] * Wq1[t * EE + k];
  q1v[t] = acc;
  __syncthreads();
  if (t < RG) {
    float a2 = breg[t];
    for (int j = 0; j < EE; ++j) a2 += q1v[j] * Wreg[t * (2 * EE) + j];
    q1reg[b * RG + t] = a2;
  }
}

// ---- head: q2 -> concat(Wreg half) -> relu -> Wq dot -----------------------
__global__ void __launch_bounds__(256)
k_head(const float* __restrict__ mu, const float* __restrict__ Wq2,
       const float* __restrict__ bq2, const float* __restrict__ Wreg,
       const float* __restrict__ q1reg, const float* __restrict__ Wq,
       const float* __restrict__ bq, float* __restrict__ out) {
  __shared__ float WTq2[EE * EE];   // 16KB
  __shared__ float Wr2[EE * RG];    //  8KB: Wr2[j*RG+r] = Wreg[r][64+j]
  __shared__ float muS[64 * EE];    // 16KB
  __shared__ float q2S[64 * EE];    // 16KB
  __shared__ float qrS[64 * RG];    //  8KB
  __shared__ float q1r[RG], wqS[RG];
  const int tid   = threadIdx.x;
  const int rbase = blockIdx.x * 64;
  const int b     = rbase >> 11;
  for (int idx = tid; idx < EE * EE; idx += 256) {
    const int e = idx >> 6, k = idx & 63;
    WTq2[k * EE + e] = Wq2[idx];
  }
  for (int idx = tid; idx < EE * RG; idx += 256) {
    const int j = idx >> 5, r = idx & 31;
    Wr2[idx] = Wreg[r * (2 * EE) + EE + j];
  }
  for (int idx = tid; idx < 64 * EE; idx += 256)
    muS[idx] = mu[(size_t)rbase * EE + idx];
  if (tid < RG) { q1r[tid] = q1reg[b * RG + tid]; wqS[tid] = Wq[tid]; }
  __syncthreads();
  const int e = tid & 63, r0 = tid >> 6;
  const float b2 = bq2[e];
  for (int p = 0; p < 16; ++p) {
    const int rl = r0 + 4 * p;
    float acc = b2;
#pragma unroll
    for (int k = 0; k < EE; ++k) acc += muS[rl * EE + k] * WTq2[k * EE + e];
    q2S[rl * EE + e] = acc;                       // q2 has no relu
  }
  __syncthreads();
  for (int idx = tid; idx < 64 * RG; idx += 256) {
    const int rl = idx >> 5, r = idx & 31;
    float acc = q1r[r];
#pragma unroll
    for (int j = 0; j < EE; ++j) acc += q2S[rl * EE + j] * Wr2[j * RG + r];
    qrS[idx] = relu_(acc);
  }
  __syncthreads();
  if (tid < 64) {
    float s = bq[0];
#pragma unroll
    for (int r = 0; r < RG; ++r) s += qrS[tid * RG + r] * wqS[r];
    out[rbase + tid] = s;
  }
}

// ---------------------------------------------------------------------------
extern "C" void kernel_launch(void* const* d_in, const int* in_sizes, int n_in,
                              void* d_out, int out_size, void* d_ws,
                              size_t ws_size, hipStream_t stream) {
  (void)in_sizes; (void)n_in; (void)out_size; (void)ws_size;
  const float* xv     = (const float*)d_in[0];
  const float* adj    = (const float*)d_in[1];
  const float* mu1    = (const float*)d_in[2];
  const float* W_mu2  = (const float*)d_in[3];
  const float* b_mu2  = (const float*)d_in[4];
  const float* W_pre  = (const float*)d_in[5];
  const float* b_pre  = (const float*)d_in[6];
  const float* W_post = (const float*)d_in[7];
  const float* b_post = (const float*)d_in[8];
  const float* Wq1    = (const float*)d_in[9];
  const float* bq1    = (const float*)d_in[10];
  const float* Wq2    = (const float*)d_in[11];
  const float* bq2    = (const float*)d_in[12];
  const float* Wreg   = (const float*)d_in[13];
  const float* breg   = (const float*)d_in[14];
  const float* Wq     = (const float*)d_in[15];
  const float* bq     = (const float*)d_in[16];

  // Workspace layout (~28.3 MB total)
  char*   ws     = (char*)d_ws;
  float*  mu     = (float*)(ws);                                   //  8 MB
  float*  mup    = (float*)(ws + (size_t)8  * 1024 * 1024);        //  8 MB
  float*  pool   = (float*)(ws + (size_t)16 * 1024 * 1024);        //  8 MB
  __bf16* mupT   = (__bf16*)(ws + (size_t)24 * 1024 * 1024);       //  4 MB
  float*  invD   = (float*)(ws + (size_t)28 * 1024 * 1024);        // 128 KB
  float*  pooled = (float*)(ws + (size_t)28 * 1024 * 1024 + 131072);
  float*  q1reg  = (float*)(ws + (size_t)28 * 1024 * 1024 + 131072 + 4096);
  float*  out    = (float*)d_out;

  k_colsum_inv<<<dim3(NN / 256, NB), 256, 0, stream>>>(adj, invD);
  k_mu_init<<<(NB * NN * EE) / 256, 256, 0, stream>>>(xv, mu1, mu);
  for (int t = 1; t < 4; ++t) {
    k_pre<<<NB * NN / 64, 256, 0, stream>>>(mu, W_pre, b_pre, mup, mupT);
    k_pool_wmma<<<dim3(NN / 128, NB), 256, 0, stream>>>(adj, invD, mupT, mup, pool);
    k_post<<<NB * NN / 64, 256, 0, stream>>>(pool, xv, mu1, W_post, b_post,
                                             W_mu2, b_mu2, mu);
  }
  k_pooled<<<NB, 256, 0, stream>>>(xv, mu, pooled);
  k_q1reg<<<NB, 64, 0, stream>>>(pooled, Wq1, bq1, Wreg, breg, q1reg);
  k_head<<<NB * NN / 64, 256, 0, stream>>>(mu, Wq2, bq2, Wreg, q1reg, Wq, bq, out);
}
